// MultiQueryAttention_19164144075512
// MI455X (gfx1250) — matrix operand order repaired and checked by reading, not compiled, not run on hardware
//
#include <hip/hip_runtime.h>

#define BATCH  4
#define SEQ    2048
#define DMODEL 1024
#define NHEADS 16
#define HDIM   64

typedef __attribute__((ext_vector_type(16))) __bf16 v16bf;
typedef __attribute__((ext_vector_type(8)))  __bf16 v8bf;
typedef __attribute__((ext_vector_type(8)))  float  v8f;

__device__ __forceinline__ __bf16 f2bf(float f) {
  unsigned u = __builtin_bit_cast(unsigned, f);
  u += 0x7FFFu + ((u >> 16) & 1u);            // round-to-nearest-even
  unsigned short h = (unsigned short)(u >> 16);
  return __builtin_bit_cast(__bf16, h);
}

__device__ __forceinline__ v16bf ld8x2(const __bf16* p0, const __bf16* p1) {
  v8bf lo = *reinterpret_cast<const v8bf*>(p0);
  v8bf hi = *reinterpret_cast<const v8bf*>(p1);
  v16bf r;
#pragma unroll
  for (int i = 0; i < 8; ++i) { r[i] = lo[i]; r[i + 8] = hi[i]; }
  return r;
}

__device__ __forceinline__ v16bf ld16(const __bf16* p) { return ld8x2(p, p + 8); }

__device__ __forceinline__ v8f wmma_bf16(v16bf a, v16bf b, v8f c) {
  return __builtin_amdgcn_wmma_f32_16x16x32_bf16(false, a, false, b, (short)0, c,
                                                 false, false);
}

// ---------- elementwise pre-passes ----------
__global__ void k_f32_to_bf16(const float* __restrict__ in, __bf16* __restrict__ out, int n) {
  int i = blockIdx.x * blockDim.x + threadIdx.x;
  if (i < n) out[i] = f2bf(in[i]);
}

// W[K][N] f32  ->  Wt[N][K] bf16
__global__ void k_transpose_w(const float* __restrict__ in, __bf16* __restrict__ out,
                              int K, int N) {
  int i = blockIdx.x * blockDim.x + threadIdx.x;
  if (i >= K * N) return;
  int k = i / N, n = i % N;
  out[(size_t)n * K + k] = f2bf(in[i]);
}

// V[B][S][HDIM] bf16 -> Vt[B][HDIM][S] bf16
__global__ void k_transpose_v(const __bf16* __restrict__ in, __bf16* __restrict__ out) {
  int i = blockIdx.x * blockDim.x + threadIdx.x;
  if (i >= BATCH * SEQ * HDIM) return;
  int d = i % HDIM;
  int s = (i / HDIM) % SEQ;
  int b = i / (HDIM * SEQ);
  out[((size_t)b * HDIM + d) * SEQ + s] = in[i];
}

// ---------- WMMA GEMM: C[M,N] = A[M,K](bf16) @ Bt[N,K](bf16)^T + bias ----------
// 1 wave per block; each wave computes a 32x64 tile (2x4 WMMA tiles,
// 8 f32 accumulators = 64 VGPRs -> no spills, 2-3 waves/SIMD occupancy).
// Per 32-k step: 4 A-loads + 8 B-loads (b128) feed 8 WMMAs (1.5 loads/WMMA).
__global__ void k_gemm_wmma(const __bf16* __restrict__ A, const __bf16* __restrict__ Bt,
                            const float* __restrict__ bias, void* __restrict__ Cout,
                            int M, int N, int K, int out_f32) {
  const int lane = threadIdx.x;
  const int col  = lane & 15;
  const int half = lane >> 4;
  const int m0 = blockIdx.x * 32;
  const int n0 = blockIdx.y * 64;

  v8f acc[2][4] = {};
  const __bf16* arow[2];
  const __bf16* brow[4];
#pragma unroll
  for (int t = 0; t < 2; ++t)
    arow[t] = A + (size_t)(m0 + t * 16 + col) * K;
#pragma unroll
  for (int u = 0; u < 4; ++u)
    brow[u] = Bt + (size_t)(n0 + u * 16 + col) * K;

  for (int k0 = 0; k0 < K; k0 += 32) {
    v16bf va0 = ld8x2(arow[0] + k0 + half * 8, arow[0] + k0 + 16 + half * 8);
    v16bf va1 = ld8x2(arow[1] + k0 + half * 8, arow[1] + k0 + 16 + half * 8);
#pragma unroll
    for (int u = 0; u < 4; ++u) {
      v16bf vb = ld8x2(brow[u] + k0 + half * 8, brow[u] + k0 + 16 + half * 8);   // transient B-frag
      acc[0][u] = wmma_bf16(va0, vb, acc[0][u]);
      acc[1][u] = wmma_bf16(va1, vb, acc[1][u]);
    }
  }

#pragma unroll
  for (int u = 0; u < 4; ++u) {
    int n = n0 + u * 16 + col;
    float bs = bias ? bias[n] : 0.0f;
#pragma unroll
    for (int t = 0; t < 2; ++t) {
#pragma unroll
      for (int r = 0; r < 8; ++r) {
        int m = m0 + t * 16 + r + half * 8;        // C layout: row = r + 8*(lane/16)
        float v = acc[t][u][r] + bs;
        if (out_f32) ((float*)Cout)[(size_t)m * N + n] = v;
        else         ((__bf16*)Cout)[(size_t)m * N + n] = f2bf(v);
      }
    }
  }
}

// ---------- causal MQA flash-attention core ----------
// grid = (SEQ/32, NHEADS, BATCH), block = 32 (one wave per 32-query tile).
// Two 16-query column groups share the K A-frags and V B-frags.
// Computes S^T = K @ Q^T so the score C-fragments reinterpret directly as the
// A-fragment of P for P@V (no LDS / cross-lane transpose needed).
__global__ void k_mqa_attn(const __bf16* __restrict__ Qb,   // [B,S,DMODEL]
                           const __bf16* __restrict__ Kb,   // [B,S,HDIM]
                           const __bf16* __restrict__ Vt,   // [B,HDIM,S]
                           __bf16* __restrict__ Ob) {       // [B,S,DMODEL]
  const int lane = threadIdx.x;
  const int col  = lane & 15;       // query-within-group for score frags
  const int half = lane >> 4;
  const int qt = blockIdx.x;
  const int h  = blockIdx.y;
  const int b  = blockIdx.z;
  const int q0 = qt * 32;

  // B-frags of Q^T (col = query, k = d): loaded once per column group.
  v16bf bq[2][2];
#pragma unroll
  for (int cg = 0; cg < 2; ++cg) {
    const __bf16* qbase =
        Qb + ((size_t)(b * SEQ + q0 + cg * 16 + col)) * DMODEL + h * HDIM;
    bq[cg][0] = ld8x2(qbase + half * 8, qbase + 16 + half * 8);        // d = 0..31
    bq[cg][1] = ld8x2(qbase + 32 + half * 8, qbase + 48 + half * 8);   // d = 32..63
  }

  v8f acc[2][4] = {};
  float m_run[2] = {-INFINITY, -INFINITY};
  float l_run[2] = {0.0f, 0.0f};

  const int ntiles = qt + 1;                    // causal: keys 0..q0+31

  for (int j = 0; j < ntiles; ++j) {
    const int k0g = j * 32;

    // A-frags of K (row = key, k = d), shared by both column groups
    const __bf16* kr0 = Kb + ((size_t)(b * SEQ + k0g + col)) * HDIM;        // keys +0..15
    const __bf16* kr1 = Kb + ((size_t)(b * SEQ + k0g + 16 + col)) * HDIM;   // keys +16..31
    v16bf a00 = ld8x2(kr0 + half * 8,      kr0 + 16 + half * 8);   // d 0..31
    v16bf a01 = ld8x2(kr0 + 32 + half * 8, kr0 + 48 + half * 8);   // d 32..63
    v16bf a10 = ld8x2(kr1 + half * 8,      kr1 + 16 + half * 8);
    v16bf a11 = ld8x2(kr1 + 32 + half * 8, kr1 + 48 + half * 8);

    v16bf pa[2];                                // A-frags of P (16q x 32k each)
#pragma unroll
    for (int cg = 0; cg < 2; ++cg) {
      v8f st0 = {}, st1 = {};
      st0 = wmma_bf16(a00, bq[cg][0], st0);
      st0 = wmma_bf16(a01, bq[cg][1], st0);
      st1 = wmma_bf16(a10, bq[cg][0], st1);
      st1 = wmma_bf16(a11, bq[cg][1], st1);

      const int query = q0 + cg * 16 + col;
      // scale (1/sqrt(Dh)=0.125 after-mask semantics) + causal mask + tile max
      float p0[8], p1[8];
      float tmax = -INFINITY;
#pragma unroll
      for (int r = 0; r < 8; ++r) {
        int key0 = k0g + r + half * 8;          // S^T row = key
        float s0 = (key0 > query) ? -INFINITY : st0[r] * 0.125f;
        float s1 = (key0 + 16 > query) ? -INFINITY : st1[r] * 0.125f;
        p0[r] = s0; p1[r] = s1;
        tmax = fmaxf(tmax, fmaxf(s0, s1));
      }
      tmax = fmaxf(tmax, __shfl_xor(tmax, 16, 32));   // combine both key halves
      float mnew  = fmaxf(m_run[cg], tmax);
      float alpha = __expf(m_run[cg] - mnew);         // tile 0 always has valid keys

      float psum = 0.0f;
#pragma unroll
      for (int r = 0; r < 8; ++r) {
        float e0 = __expf(p0[r] - mnew);
        float e1 = __expf(p1[r] - mnew);
        psum += e0 + e1;
        pa[cg][r]     = f2bf(e0);   // k = r + 8*half       (keys 0..15)
        pa[cg][r + 8] = f2bf(e1);   // k = 16 + r + 8*half  (keys 16..31)
      }
      psum += __shfl_xor(psum, 16, 32);
      l_run[cg] = l_run[cg] * alpha + psum;
      m_run[cg] = mnew;

      // rescale O accumulator; O rows = query r + 8*half, alpha lives at lane==query
#pragma unroll
      for (int r = 0; r < 8; ++r) {
        float ar = __shfl(alpha, r + half * 8, 32);
        acc[cg][0][r] *= ar; acc[cg][1][r] *= ar;
        acc[cg][2][r] *= ar; acc[cg][3][r] *= ar;
      }
    }

    // P @ V : B-frag lane holds column d, keys 8*half+0..7 and 16+8*half+0..7; shared by both groups
    const __bf16* vb = Vt + ((size_t)b * HDIM) * SEQ;
#pragma unroll
    for (int t = 0; t < 4; ++t) {
      v16bf bv = ld8x2(vb + (size_t)(t * 16 + col) * SEQ + k0g + half * 8,
                       vb + (size_t)(t * 16 + col) * SEQ + k0g + 16 + half * 8);
      acc[0][t] = wmma_bf16(pa[0], bv, acc[0][t]);
      acc[1][t] = wmma_bf16(pa[1], bv, acc[1][t]);
    }
  }

  // epilogue: divide by l (per query row), write bf16 [B,S,H*Dh]
#pragma unroll
  for (int cg = 0; cg < 2; ++cg) {
#pragma unroll
    for (int r = 0; r < 8; ++r) {
      float lr  = __shfl(l_run[cg], r + half * 8, 32);
      float rcp = 1.0f / lr;                    // >0: diagonal always unmasked
      int rowq = q0 + cg * 16 + r + half * 8;
      __bf16* op = Ob + ((size_t)(b * SEQ + rowq)) * DMODEL + h * HDIM;
      op[0 * 16 + col] = f2bf(acc[cg][0][r] * rcp);
      op[1 * 16 + col] = f2bf(acc[cg][1][r] * rcp);
      op[2 * 16 + col] = f2bf(acc[cg][2][r] * rcp);
      op[3 * 16 + col] = f2bf(acc[cg][3][r] * rcp);
    }
  }
}

// ---------- host launch ----------
extern "C" void kernel_launch(void* const* d_in, const int* in_sizes, int n_in,
                              void* d_out, int out_size, void* d_ws, size_t ws_size,
                              hipStream_t stream) {
  (void)in_sizes; (void)n_in; (void)out_size; (void)ws_size;
  const float* x  = (const float*)d_in[0];
  const float* Wq = (const float*)d_in[1];
  const float* bq = (const float*)d_in[2];
  const float* Wk = (const float*)d_in[3];
  const float* bk = (const float*)d_in[4];
  const float* Wv = (const float*)d_in[5];
  const float* bv = (const float*)d_in[6];
  const float* Wo = (const float*)d_in[7];
  const float* bo = (const float*)d_in[8];
  float* out = (float*)d_out;

  const int M = BATCH * SEQ;     // 8192
  char* ws = (char*)d_ws;
  size_t off = 0;
  auto carve = [&](size_t elems) {
    __bf16* p = (__bf16*)(ws + off);
    off = (off + elems * sizeof(__bf16) + 255) & ~(size_t)255;
    return p;
  };
  __bf16* xb  = carve((size_t)M * DMODEL);
  __bf16* Wqt = carve((size_t)DMODEL * DMODEL);
  __bf16* Wkt = carve((size_t)HDIM * DMODEL);
  __bf16* Wvt = carve((size_t)HDIM * DMODEL);
  __bf16* Wot = carve((size_t)DMODEL * DMODEL);
  __bf16* Qb  = carve((size_t)M * DMODEL);
  __bf16* Kb  = carve((size_t)M * HDIM);
  __bf16* Vb  = carve((size_t)M * HDIM);
  __bf16* Vt  = carve((size_t)M * HDIM);
  __bf16* Ob  = carve((size_t)M * DMODEL);

  const int EB = 256;
  int nx = M * DMODEL;
  k_f32_to_bf16<<<(nx + EB - 1) / EB, EB, 0, stream>>>(x, xb, nx);
  int nw = DMODEL * DMODEL;
  k_transpose_w<<<(nw + EB - 1) / EB, EB, 0, stream>>>(Wq, Wqt, DMODEL, DMODEL);
  int nkv = DMODEL * HDIM;
  k_transpose_w<<<(nkv + EB - 1) / EB, EB, 0, stream>>>(Wk, Wkt, DMODEL, HDIM);
  k_transpose_w<<<(nkv + EB - 1) / EB, EB, 0, stream>>>(Wv, Wvt, DMODEL, HDIM);
  k_transpose_w<<<(nw + EB - 1) / EB, EB, 0, stream>>>(Wo, Wot, DMODEL, DMODEL);

  // projections (bf16 outputs)
  k_gemm_wmma<<<dim3(M / 32, DMODEL / 64), 32, 0, stream>>>(xb, Wqt, bq, Qb,
                                                            M, DMODEL, DMODEL, 0);
  k_gemm_wmma<<<dim3(M / 32, HDIM / 64), 32, 0, stream>>>(xb, Wkt, bk, Kb,
                                                          M, HDIM, DMODEL, 0);
  k_gemm_wmma<<<dim3(M / 32, HDIM / 64), 32, 0, stream>>>(xb, Wvt, bv, Vb,
                                                          M, HDIM, DMODEL, 0);
  int nv = BATCH * SEQ * HDIM;
  k_transpose_v<<<(nv + EB - 1) / EB, EB, 0, stream>>>(Vb, Vt);

  // attention
  k_mqa_attn<<<dim3(SEQ / 32, NHEADS, BATCH), 32, 0, stream>>>(Qb, Kb, Vt, Ob);

  // output projection (f32 output)
  k_gemm_wmma<<<dim3(M / 32, DMODEL / 64), 32, 0, stream>>>(Ob, Wot, bo, out,
                                                            M, DMODEL, DMODEL, 1);
}
